// CodeEncoder_26139170963539
// MI455X (gfx1250) — compile-verified, matrix operand order checked by reading, not run on hardware
//
#include <hip/hip_runtime.h>
#include <hip/hip_bf16.h>
#include <math.h>

// ---------- types ----------
typedef __attribute__((ext_vector_type(16))) __bf16 v16bf;
typedef __attribute__((ext_vector_type(8)))  __bf16 v8bf;
typedef __attribute__((ext_vector_type(4)))  __bf16 v4bf;
typedef __attribute__((ext_vector_type(8)))  float  v8f;

// problem constants (reference: B=64, V=50, C=64, D=256, H=128)
#define CROW 64
#define DEMB 256
#define HDIM 128

// ------------------------------------------------------------------
// Kernel 0: pack W1 [128,256] f32 -> bf16 in WMMA B-fragment order.
// B is K x N = 32 x 16 per fragment, B[k][n] = W1[n][k].
// 16-bit B layout (mirrors A layout, N across lanes):
//   lane l: n = l%16, base_k = (l>=16 ? 8 : 0)
//   elems 0..7  -> K = kbase + base_k + 0..7
//   elems 8..15 -> K = kbase + 16 + base_k + 0..7
// Fragment index = (n_tile*8 + k_step)*32 + lane, 16 bf16 each.
// ------------------------------------------------------------------
__global__ void pack_w1_kernel(const float* __restrict__ W1,
                               __bf16* __restrict__ w1p) {
    int idx = blockIdx.x * blockDim.x + threadIdx.x;   // 0 .. 2047
    if (idx >= 8 * 8 * 32) return;
    int lane  = idx & 31;
    int ks    = (idx >> 5) & 7;
    int nt    = idx >> 8;
    int lmod  = lane & 15;
    int lhalf = lane >> 4;
    int n     = nt * 16 + lmod;
    int kb    = ks * 32 + lhalf * 8;
    const float* src = W1 + (size_t)n * DEMB;
    __bf16* dst = w1p + (size_t)idx * 16;
#pragma unroll
    for (int j = 0; j < 8; ++j) dst[j]     = (__bf16)src[kb + j];
#pragma unroll
    for (int j = 0; j < 8; ++j) dst[8 + j] = (__bf16)src[kb + 16 + j];
}

// ------------------------------------------------------------------
// Main kernel: one workgroup (256 thr = 8 waves) per (b,v) row.
// ------------------------------------------------------------------
__global__ __launch_bounds__(256) void code_encoder_kernel(
    const int*   __restrict__ input_code,   // [3200,64]
    const int*   __restrict__ length_code,  // [3200]
    const float* __restrict__ embed_table,  // [20000,256]
    const float* __restrict__ b1,           // [128]
    const float* __restrict__ W2,           // [1,128]
    const float* __restrict__ b2,           // [1]
    const __bf16* __restrict__ w1p,         // packed B fragments
    float*       __restrict__ out)          // [3200,256]
{
    extern __shared__ char smem[];
    float*  embf  = (float*)smem;                          // 64*256 f32   (64 KB)
    __bf16* embA  = (__bf16*)(smem + 65536);               // A-fragments  (32 KB)
    float*  wpart = (float*)(smem + 65536 + 32768);        // 8*64 f32     ( 2 KB)
    float*  attnS = wpart + 512;                           // 64 f32
    int*    codesS = (int*)(attnS + 64);                   // 64 int

    const int row  = blockIdx.x;
    const int t    = threadIdx.x;
    const int lane = t & 31;
    const int wave = t >> 5;

    if (t < CROW) codesS[t] = input_code[row * CROW + t];
    __syncthreads();

    // ---- gather: f32 copy + bf16 copy written directly in A-fragment order
    // A fragment (16x32 bf16): lane l: m = l%16, base_k = (l>=16?8:0)
    //   elems 0..7  -> k' = base_k + 0..7 ; elems 8..15 -> k' = 16 + base_k + 0..7
#pragma unroll
    for (int i = 0; i < 16; ++i) {
        int idx = i * 256 + t;          // 0..4095 float4 slots
        int c   = idx >> 6;             // code index 0..63
        int d4  = idx & 63;             // float4 index within row
        const float4 v = ((const float4*)(embed_table + (size_t)codesS[c] * DEMB))[d4];
        ((float4*)(embf + c * DEMB))[d4] = v;

        // fragment coordinates for the 4 consecutive d = d4*4 .. d4*4+3
        int mt    = c >> 4;
        int lmodc = c & 15;
        int ks    = d4 >> 3;                    // d/32
        int kp    = (d4 & 7) * 4;               // k' within 32 (multiple of 4)
        int lhalf = (kp >> 3) & 1;
        int jbase = (kp & 7) + ((kp >> 4) & 1) * 8;
        int frag  = (mt * 8 + ks) * 32 + (lmodc + 16 * lhalf);
        v4bf bv;
        bv.x = (__bf16)v.x; bv.y = (__bf16)v.y;
        bv.z = (__bf16)v.z; bv.w = (__bf16)v.w;
        *(v4bf*)(embA + frag * 16 + jbase) = bv;
    }
    __syncthreads();

    // ---- WMMA: h = emb[64,256] @ W1^T[256,128]; wave nt owns 16 columns
    const int nt    = wave;
    const int lmod  = lane & 15;
    const int lhalf = lane >> 4;

    v8f acc[4];
#pragma unroll
    for (int mt = 0; mt < 4; ++mt) acc[mt] = (v8f){0.f,0.f,0.f,0.f,0.f,0.f,0.f,0.f};

#pragma unroll
    for (int ks = 0; ks < 8; ++ks) {
        const v16bf bfrag = *((const v16bf*)w1p + ((nt * 8 + ks) * 32 + lane));
#pragma unroll
        for (int mt = 0; mt < 4; ++mt) {
            const v16bf afrag = *((const v16bf*)embA + ((mt * 8 + ks) * 32 + lane));
            acc[mt] = __builtin_amdgcn_wmma_f32_16x16x32_bf16(
                false, afrag, false, bfrag, (short)0, acc[mt], false, false);
        }
    }

    // ---- tanh + dot with W2, reduced over the 16 N-lanes of this wave
    // C/D layout: VGPR r -> M = r (lanes 0-15) / 8+r (lanes 16-31), N = lane%16
    const float b1n = b1[nt * 16 + lmod];
    const float w2n = W2[nt * 16 + lmod];
#pragma unroll
    for (int mt = 0; mt < 4; ++mt) {
#pragma unroll
        for (int r = 0; r < 8; ++r) {
            float h = tanhf(acc[mt][r] + b1n);
            float p = h * w2n;
            p += __shfl_xor(p, 1, 32);
            p += __shfl_xor(p, 2, 32);
            p += __shfl_xor(p, 4, 32);
            p += __shfl_xor(p, 8, 32);
            if (lmod == 0)
                wpart[wave * 64 + mt * 16 + lhalf * 8 + r] = p;
        }
    }
    __syncthreads();

    // ---- masked softmax over the 64 codes (wave 0; deterministic order)
    if (wave == 0) {
        const int   len = length_code[row];
        const float b2v = b2[0];
        float v0 = b2v, v1 = b2v;
#pragma unroll
        for (int w = 0; w < 8; ++w) {
            v0 += wpart[w * 64 + lane];
            v1 += wpart[w * 64 + lane + 32];
        }
        if (lane      >= len) v0 = -1e9f;
        if (lane + 32 >= len) v1 = -1e9f;
        float m = fmaxf(v0, v1);
        m = fmaxf(m, __shfl_xor(m, 1, 32));
        m = fmaxf(m, __shfl_xor(m, 2, 32));
        m = fmaxf(m, __shfl_xor(m, 4, 32));
        m = fmaxf(m, __shfl_xor(m, 8, 32));
        m = fmaxf(m, __shfl_xor(m, 16, 32));
        float e0 = expf(v0 - m), e1 = expf(v1 - m);
        float s = e0 + e1;
        s += __shfl_xor(s, 1, 32);
        s += __shfl_xor(s, 2, 32);
        s += __shfl_xor(s, 4, 32);
        s += __shfl_xor(s, 8, 32);
        s += __shfl_xor(s, 16, 32);
        float inv = 1.0f / s;
        attnS[lane]      = e0 * inv;
        attnS[lane + 32] = e1 * inv;
    }
    __syncthreads();

    // ---- out[d] = sum_c attn[c] * emb_f32[c][d]  (thread t = column d)
    float o = 0.f;
#pragma unroll 8
    for (int c = 0; c < CROW; ++c)
        o = fmaf(attnS[c], embf[c * DEMB + t], o);
    out[(size_t)row * DEMB + t] = o;
}

// ------------------------------------------------------------------
extern "C" void kernel_launch(void* const* d_in, const int* in_sizes, int n_in,
                              void* d_out, int out_size, void* d_ws, size_t ws_size,
                              hipStream_t stream) {
    const int*   input_code  = (const int*)d_in[0];
    const int*   length_code = (const int*)d_in[1];
    const float* embed_table = (const float*)d_in[2];
    const float* W1          = (const float*)d_in[3];
    const float* b1          = (const float*)d_in[4];
    const float* W2          = (const float*)d_in[5];
    const float* b2          = (const float*)d_in[6];
    float*  out = (float*)d_out;
    __bf16* w1p = (__bf16*)d_ws;     // 8*8*32*16 bf16 = 64 KB

    // pack W1 into bf16 B-fragments (runs each launch; deterministic)
    pack_w1_kernel<<<8, 256, 0, stream>>>(W1, w1p);

    const int rows = in_sizes[1];    // B*V = 3200
    const size_t smem = 65536 + 32768 + 2048 + 256 + 256;  // ~100 KB dynamic LDS
    code_encoder_kernel<<<rows, 256, smem, stream>>>(
        input_code, length_code, embed_table, b1, W2, b2, w1p, out);
}
